// StyledConv_33861522162120
// MI455X (gfx1250) — matrix operand order repaired and checked by reading, not compiled, not run on hardware
//
#include <hip/hip_runtime.h>

typedef __attribute__((ext_vector_type(16))) __bf16 v16bf;
typedef __attribute__((ext_vector_type(8)))  float  v8f;
typedef __attribute__((ext_vector_type(4)))  unsigned int v4u;   // 16B aligned

#define CH 64        // Cin == Cout
#define HH 256
#define WW 256
#define HT 8         // output rows per block
#define WT 16        // output cols per block
#define RIN 10       // HT+2 (halo)
#define CINT 18      // WT+2 (halo)
#define NPIX (RIN*CINT)          // 180 staged pixels
#define PIXS 72                  // padded halves per pixel -> 144B, 16B-aligned fragments
#define FRAG_PER_B 36864         // 9k * 2kc * 4mt * 32lane * 16e  (= 64*64*9)

// ---------------- Kernel 1: modulate + demodulate + swizzle weights ----------------
// Writes bf16 hi/lo weight planes directly in the V_WMMA_F32_16X16X32_BF16
// A operand in this code's contiguous k order (same as B): lane = m + (K>>4)*16 ; elem e = K&15
__global__ void weight_prep(const float* __restrict__ style,
                            const float* __restrict__ weight,
                            const float* __restrict__ mod_w,
                            const float* __restrict__ mod_b,
                            unsigned short* __restrict__ wAhi_u,
                            unsigned short* __restrict__ wAlo_u) {
    __shared__ float sS[CH];
    __shared__ float sD[CH];
    const int b = blockIdx.x, t = threadIdx.x;
    __bf16* wAhi = (__bf16*)wAhi_u;
    __bf16* wAlo = (__bf16*)wAlo_u;

    if (t < CH) {                       // s = style @ (mod_w * lin_scale) + mod_b
        float a = 0.f;
        for (int j = 0; j < 512; ++j)
            a += style[b * 512 + j] * mod_w[j * CH + t];
        sS[t] = a * 0.044194173824159216f /* 1/sqrt(512) */ + mod_b[t];
    }
    __syncthreads();
    if (t < CH) {                       // demod over (ci, kh, kw)
        float sum = 0.f;
        const float* wr = weight + t * CH * 9;
        for (int ci = 0; ci < CH; ++ci) {
            float sc = sS[ci] * (1.0f / 24.0f);   // conv_scale = 1/sqrt(64*9)
            for (int kk = 0; kk < 9; ++kk) {
                float m = wr[ci * 9 + kk] * sc;
                sum += m * m;
            }
        }
        sD[t] = 1.0f / sqrtf(sum + 1e-8f);
    }
    __syncthreads();
    // emit swizzled A fragments (hi + lo split)
    for (int i = t; i < FRAG_PER_B; i += 256) {
        int e    = i & 15;
        int lane = (i >> 4) & 31;
        int mt   = (i >> 9) & 3;
        int kc   = (i >> 11) & 1;
        int k    = i >> 12;                       // 0..8
        int m    = lane & 15;
        int K    = (e & 15) | ((lane >> 4) << 4);
        int ci   = kc * 32 + K;
        int co   = mt * 16 + m;
        float v  = (1.0f / 24.0f) * weight[(co * CH + ci) * 9 + k] * sS[ci] * sD[co];
        __bf16 hi = (__bf16)v;
        float  lo = v - (float)hi;
        size_t o  = (size_t)b * FRAG_PER_B + i;
        wAhi[o] = hi;
        wAlo[o] = (__bf16)lo;
    }
}

// ---------------- Kernel 2: implicit-GEMM 3x3 conv via bf16x3 WMMA ----------------
__global__ __launch_bounds__(256)
void styled_conv_main(const float* __restrict__ x,
                      const float* __restrict__ noise,
                      const float* __restrict__ nwp,
                      const float* __restrict__ bias,
                      const unsigned short* __restrict__ wAhi_u,
                      const unsigned short* __restrict__ wAlo_u,
                      float* __restrict__ out) {
    __shared__ __bf16 sHi[NPIX * PIXS];
    __shared__ __bf16 sLo[NPIX * PIXS];

    const int tw = blockIdx.x, th = blockIdx.y, b = blockIdx.z;
    const int w0 = tw * WT, h0 = th * HT;
    const int tid = threadIdx.x;

    // ---- stage x halo tile to LDS, transposed to [pixel][ci], bf16 hi/lo split ----
    for (int i = tid; i < CH * NPIX; i += 256) {
        int ci  = i / NPIX;
        int pix = i - ci * NPIX;
        int row = pix / CINT;
        int col = pix - row * CINT;
        int h = h0 - 1 + row, w = w0 - 1 + col;
        float v = 0.f;
        if ((unsigned)h < (unsigned)HH && (unsigned)w < (unsigned)WW)
            v = x[(((size_t)(b * CH + ci) * HH) + h) * WW + w];
        __bf16 hi = (__bf16)v;
        float  lo = v - (float)hi;
        sHi[pix * PIXS + ci] = hi;
        sLo[pix * PIXS + ci] = (__bf16)lo;
    }
    __syncthreads();

    const int wid  = tid >> 5;
    const int lane = tid & 31;
    const int mt   = wid & 3;            // co tile (16 channels)
    const int rgrp = (wid >> 2) * 4;     // this wave's 4 output rows
    const int lh   = lane >> 4;          // lane half
    const int n    = lane & 15;          // pixel within 16-wide segment

    v8f zero = {0.f, 0.f, 0.f, 0.f, 0.f, 0.f, 0.f, 0.f};
    v8f acc[4] = {zero, zero, zero, zero};

    const char* cHi = (const char*)sHi;
    const char* cLo = (const char*)sLo;
    const __bf16* pAhi = (const __bf16*)wAhi_u;
    const __bf16* pAlo = (const __bf16*)wAlo_u;

    union BU { v4u q[2]; v16bf v; };

    for (int kh = 0; kh < 3; ++kh) {
        for (int kw = 0; kw < 3; ++kw) {
            const int k = kh * 3 + kw;
#pragma unroll
            for (int kc = 0; kc < 2; ++kc) {
                const int fo = ((((b * 9 + k) * 2 + kc) * 4 + mt) * 32 + lane) * 16;
                v16bf ahi = *(const v16bf*)(pAhi + fo);   // 32B/lane, coalesced b128 pair
                v16bf alo = *(const v16bf*)(pAlo + fo);
#pragma unroll
                for (int r = 0; r < 4; ++r) {
                    const int ro  = rgrp + r;
                    const int pix = (ro + kh) * CINT + (n + kw);
                    // byte offset of this lane's 32B B-fragment: 16B aligned by construction
                    const int bo  = pix * (PIXS * 2) + kc * 64 + lh * 32;
                    BU bh, bl;
                    bh.q[0] = *(const v4u*)(cHi + bo);
                    bh.q[1] = *(const v4u*)(cHi + bo + 16);
                    bl.q[0] = *(const v4u*)(cLo + bo);
                    bl.q[1] = *(const v4u*)(cLo + bo + 16);
                    // D = A*B + C  (bf16x3: hh + hl + lh)
                    acc[r] = __builtin_amdgcn_wmma_f32_16x16x32_bf16(
                        false, ahi, false, bh.v, (short)0, acc[r], false, false);
                    acc[r] = __builtin_amdgcn_wmma_f32_16x16x32_bf16(
                        false, ahi, false, bl.v, (short)0, acc[r], false, false);
                    acc[r] = __builtin_amdgcn_wmma_f32_16x16x32_bf16(
                        false, alo, false, bh.v, (short)0, acc[r], false, false);
                }
            }
        }
    }

    // ---- epilogue: + noise, + bias, leaky_relu * sqrt(2), store ----
    const float nw = nwp[0];
    float bv[8];
#pragma unroll
    for (int v = 0; v < 8; ++v) bv[v] = bias[mt * 16 + lh * 8 + v];

#pragma unroll
    for (int r = 0; r < 4; ++r) {
        const int h = h0 + rgrp + r;
        const float nz = nw * noise[((size_t)b * HH + h) * WW + w0 + n];
        float* op = out + ((size_t)(b * CH + mt * 16 + lh * 8) * HH + h) * WW + w0 + n;
#pragma unroll
        for (int v = 0; v < 8; ++v) {
            float t = acc[r][v] + nz + bv[v];
            t = (t > 0.f ? t : 0.2f * t) * 1.4142135623730951f;
            op[(size_t)v * HH * WW] = t;
        }
    }
}

extern "C" void kernel_launch(void* const* d_in, const int* in_sizes, int n_in,
                              void* d_out, int out_size, void* d_ws, size_t ws_size,
                              hipStream_t stream) {
    const float* x      = (const float*)d_in[0];
    const float* style  = (const float*)d_in[1];
    const float* noise  = (const float*)d_in[2];
    const float* weight = (const float*)d_in[3];
    const float* mod_w  = (const float*)d_in[4];
    const float* mod_b  = (const float*)d_in[5];
    const float* nw     = (const float*)d_in[6];
    const float* bias   = (const float*)d_in[7];
    float* out = (float*)d_out;

    unsigned short* wAhi = (unsigned short*)d_ws;
    unsigned short* wAlo = wAhi + (size_t)8 * FRAG_PER_B;

    weight_prep<<<8, 256, 0, stream>>>(style, weight, mod_w, mod_b, wAhi, wAlo);

    dim3 grid(WW / WT, HH / HT, 8);   // (16, 32, 8)
    styled_conv_main<<<grid, 256, 0, stream>>>(x, noise, nw, bias, wAhi, wAlo, out);
}
